// EmformerEncoder_20796231647925
// MI455X (gfx1250) — compile-verified, matrix operand order checked by reading, not run on hardware
//
#include <hip/hip_runtime.h>
#include <hip/hip_bf16.h>
#include <hip/hip_fp16.h>

// ---------------- static config (mirrors reference) ----------------
#define B_    8
#define TIN   2048
#define FIN   80
#define T_    512      // post-subsample frames
#define D_    512
#define H_    8
#define HS_   64
#define DFF_  2048
#define NL_   4
#define RTOT  112      // (N-1)*RIGHT = 7*16
#define S_    624      // RTOT + T
#define SCOLS 640      // padded score columns (S_ rounded to 32)
#define MTOT  (B_*S_)  // 4992 tokens (multiple of 64)
#define LNEPS 1e-3f

typedef __attribute__((ext_vector_type(16))) _Float16 v16h;
typedef __attribute__((ext_vector_type(8)))  _Float16 v8h;
typedef __attribute__((ext_vector_type(8)))  float    v8f;

__device__ __forceinline__ v8f vzero8() {
  v8f z;
#pragma unroll
  for (int i = 0; i < 8; ++i) z[i] = 0.0f;
  return z;
}

__device__ __forceinline__ v16h pack16(v8h lo, v8h hi) {
  v16h a;
#pragma unroll
  for (int i = 0; i < 8; ++i) { a[i] = lo[i]; a[i + 8] = hi[i]; }
  return a;
}

// =====================================================================
// 1) Input dense:  audio[B,2048,80] @ W_in[80,128] + b_in -> dh[B,2048,128]
// =====================================================================
__global__ void __launch_bounds__(256) k_dense_in(
    const float* __restrict__ audio, const float* __restrict__ Win,
    const float* __restrict__ bin, float* __restrict__ out) {
  int idx = blockIdx.x * 256 + threadIdx.x;          // B*2048*128
  if (idx >= B_ * TIN * 128) return;
  int n   = idx & 127;
  int row = idx >> 7;                                // b*2048 + t
  const float* a = audio + (size_t)row * FIN;
  float s = bin[n];
#pragma unroll 8
  for (int k = 0; k < FIN; ++k) s = fmaf(a[k], Win[k * 128 + n], s);
  out[idx] = s;
}

// =====================================================================
// 2) Emformer mask  (bytes, [S_,S_])
// =====================================================================
__global__ void __launch_bounds__(256) k_build_mask(unsigned char* __restrict__ mask) {
  int idx = blockIdx.x * 256 + threadIdx.x;
  if (idx >= S_ * S_) return;
  int q = idx / S_, k = idx % S_;
  const int CH = 64, LF = 32, RG = 16, NCH = 8;
  bool m;
  if (q < RTOT) {
    int ri = q / RG;
    if (k < RTOT) { m = (ri == (k / RG)); }
    else {
      int kk = k - RTOT;
      int lo = ri * CH - LF; if (lo < 0) lo = 0;
      int hi = (ri + 1) * CH; if (hi > T_) hi = T_;
      m = (kk >= lo && kk < hi);
    }
  } else {
    int qq = q - RTOT, qi = qq / CH;
    if (k < RTOT) { m = ((k / RG) == qi) && (qi < NCH - 1); }
    else {
      int kk = k - RTOT;
      int lo = qi * CH - LF; if (lo < 0) lo = 0;
      int hi = (qi + 1) * CH; if (hi > T_) hi = T_;
      m = (kk >= lo && kk < hi);
    }
  }
  mask[idx] = m ? 1 : 0;
}

// =====================================================================
// 3) Stack-subsample + right-context hard copy
// =====================================================================
__global__ void __launch_bounds__(256) k_gather(
    const float* __restrict__ dh, float* __restrict__ x) {
  int idx = blockIdx.x * 256 + threadIdx.x;          // MTOT*D_
  if (idx >= MTOT * D_) return;
  int d = idx & 511;
  int rem = idx >> 9;                                // b*624 + s
  int s = rem % S_, b = rem / S_;
  int tt = (s < RTOT) ? (((s >> 4) + 1) * 64 + (s & 15)) : (s - RTOT);
  x[idx] = dh[((size_t)(b * TIN + tt * 4 + (d >> 7))) * 128 + (d & 127)];
}

// =====================================================================
// 4) LayerNorm over D=512; one wave per token; emits f16 and/or f32
// =====================================================================
__global__ void __launch_bounds__(256) k_layernorm(
    const float* __restrict__ in, const float* __restrict__ g,
    const float* __restrict__ bta, _Float16* __restrict__ out16,
    float* __restrict__ out32, int ntok) {
  int wave = threadIdx.x >> 5, lane = threadIdx.x & 31;
  int token = blockIdx.x * 8 + wave;
  if (token >= ntok) return;
  const float* p = in + (size_t)token * D_;
  float v[16];
  float s = 0.0f;
#pragma unroll
  for (int i = 0; i < 16; ++i) { v[i] = p[lane + 32 * i]; s += v[i]; }
#pragma unroll
  for (int off = 16; off > 0; off >>= 1) s += __shfl_xor(s, off, 32);
  float mu = s * (1.0f / 512.0f);
  float vs = 0.0f;
#pragma unroll
  for (int i = 0; i < 16; ++i) { float d = v[i] - mu; vs = fmaf(d, d, vs); }
#pragma unroll
  for (int off = 16; off > 0; off >>= 1) vs += __shfl_xor(vs, off, 32);
  float inv = rsqrtf(vs * (1.0f / 512.0f) + LNEPS);
#pragma unroll
  for (int i = 0; i < 16; ++i) {
    int c = lane + 32 * i;
    float o = (v[i] - mu) * inv * g[c] + bta[c];
    if (out16) out16[(size_t)token * D_ + c] = (_Float16)o;
    if (out32) out32[(size_t)token * D_ + c] = o;
  }
}

// =====================================================================
// 5) WMMA GEMM:  C[M,N] = A16[M,K] * W32[K,N] + bias (+resid) (relu?)
//    64x128 workgroup tile, BK=32, 8 waves (4x2), 4 accum tiles / wave.
//    A tile staged with CDNA5 async global->LDS copy (ASYNCcnt).
//    W tile f32->f16 converted and k-pair packed into b32 LDS stores,
//    stored transposed so B-fragments are contiguous b128 LDS reads.
// =====================================================================
__global__ void __launch_bounds__(256) k_gemm(
    const _Float16* __restrict__ A, const float* __restrict__ W,
    const float* __restrict__ bias, _Float16* __restrict__ out16,
    float* __restrict__ out32, const float* __restrict__ resid,
    int M, int N, int K, int relu) {
  __shared__ __align__(16) _Float16 As[64][40];
  __shared__ __align__(16) _Float16 Bt[128][40];

  const int t    = threadIdx.x;
  const int lane = t & 31;
  const int wave = t >> 5;
  const int wm   = wave >> 1;          // 0..3  (16-row band)
  const int wn   = wave & 1;           // 0..1  (64-col band)
  const int r    = lane & 15;
  const int hf   = lane >> 4;
  const int mbase = blockIdx.y * 64;
  const int nbase = blockIdx.x * 128;

  v8f acc[4];
#pragma unroll
  for (int i = 0; i < 4; ++i) acc[i] = vzero8();

  const int arow = t >> 2, acg = t & 3;             // A staging role
  for (int kb = 0; kb < K; kb += 32) {
    __syncthreads();
    // ---- stage A tile [64 x 32] f16 via async global->LDS (b128/lane) ----
    {
      const _Float16* ap = A + (size_t)(mbase + arow) * K + kb + 8 * acg;
      if (kb + 32 < K) __builtin_prefetch(ap + 32, 0, 1);
      unsigned ldsdst = (unsigned)(size_t)&As[arow][8 * acg];  // low 32 bits = LDS offset
      unsigned long long ga = (unsigned long long)(size_t)ap;
      asm volatile("global_load_async_to_lds_b128 %0, %1, off"
                   :: "v"(ldsdst), "v"(ga)
                   : "memory");
    }
    // ---- stage W tile [32 x 128] f32 -> f16, transposed into Bt[n][k],
    //      k-pairs packed -> ds_store_b32 ----
#pragma unroll
    for (int i = 0; i < 2; ++i) {
      int idx = t + 256 * i;                        // 0..511
      int ng = idx & 31;                            // 4-col group
      int kp = idx >> 5;                            // 0..15 (k pair)
      const float4 w0 = *(const float4*)(W + (size_t)(kb + 2 * kp)     * N + nbase + 4 * ng);
      const float4 w1 = *(const float4*)(W + (size_t)(kb + 2 * kp + 1) * N + nbase + 4 * ng);
      union { _Float16 h[2]; unsigned u; } pk;
      pk.h[0] = (_Float16)w0.x; pk.h[1] = (_Float16)w1.x;
      *(unsigned*)&Bt[4 * ng + 0][2 * kp] = pk.u;
      pk.h[0] = (_Float16)w0.y; pk.h[1] = (_Float16)w1.y;
      *(unsigned*)&Bt[4 * ng + 1][2 * kp] = pk.u;
      pk.h[0] = (_Float16)w0.z; pk.h[1] = (_Float16)w1.z;
      *(unsigned*)&Bt[4 * ng + 2][2 * kp] = pk.u;
      pk.h[0] = (_Float16)w0.w; pk.h[1] = (_Float16)w1.w;
      *(unsigned*)&Bt[4 * ng + 3][2 * kp] = pk.u;
    }
    // async A copy must land before any wave reads LDS
    asm volatile("s_wait_asynccnt 0" ::: "memory");
    __syncthreads();
    // ---- compute: one 16x16x32 WMMA per accumulator ----
    v16h afr;
    {
      v8h lo = *(const v8h*)&As[16 * wm + r][8 * hf];
      v8h hi = *(const v8h*)&As[16 * wm + r][16 + 8 * hf];
      afr = pack16(lo, hi);
    }
#pragma unroll
    for (int i = 0; i < 4; ++i) {
      int col = 64 * wn + 16 * i + r;
      v8h lo = *(const v8h*)&Bt[col][8 * hf];
      v8h hi = *(const v8h*)&Bt[col][16 + 8 * hf];
      v16h bfr = pack16(lo, hi);
      acc[i] = __builtin_amdgcn_wmma_f32_16x16x32_f16(
          false, afr, false, bfr, (short)0, acc[i], false, false);
    }
  }

  // ---- epilogue: bias (+relu) (+resid), write f16 and/or f32 ----
#pragma unroll
  for (int i = 0; i < 4; ++i) {
    int n = nbase + 64 * wn + 16 * i + r;
    float bv = bias[n];
#pragma unroll
    for (int j = 0; j < 8; ++j) {
      int m = mbase + 16 * wm + j + 8 * hf;
      float v = acc[i][j] + bv;
      if (relu) v = fmaxf(v, 0.0f);
      size_t o = (size_t)m * N + n;
      if (resid) v += resid[o];
      if (out32) out32[o] = v;
      if (out16) out16[o] = (_Float16)v;
    }
  }
}

// =====================================================================
// 6) Fused masked attention for one (b, h, 16-query tile).
//    Scores (16 x 640) live in LDS; QK^T and P*V via WMMA.
// =====================================================================
__global__ void __launch_bounds__(128) k_attention(
    const _Float16* __restrict__ q16, const _Float16* __restrict__ k16,
    const _Float16* __restrict__ v16, const unsigned char* __restrict__ mask,
    _Float16* __restrict__ out) {
  __shared__ __align__(32) float sc[16][656];   // 656 stride: 32B-aligned rows
  __shared__ float red[16][8];
  __shared__ float rowstat[16];

  const int qt = blockIdx.x;                    // 0..38
  const int h  = blockIdx.y;
  const int b  = blockIdx.z;
  const int t    = threadIdx.x;
  const int lane = t & 31;
  const int wave = t >> 5;
  const int r  = lane & 15;
  const int hf = lane >> 4;
  const int qbase = qt * 16;
  const float scale = 0.125f;                   // 1/sqrt(HS)

  // ---- Q fragments (K = 64 -> two 32-chunks), rows qbase+r ----
  const _Float16* qp = q16 + ((size_t)(b * S_ + qbase + r)) * D_ + h * HS_;
  v16h aq[2];
#pragma unroll
  for (int kc = 0; kc < 2; ++kc) {
    v8h lo = *(const v8h*)(qp + kc * 32 + 8 * hf);
    v8h hi = *(const v8h*)(qp + kc * 32 + 16 + 8 * hf);
    aq[kc] = pack16(lo, hi);
  }

  // ---- scores = Q*K^T (d is contiguous for BOTH operands) ----
  for (int kt = wave; kt < S_ / 16; kt += 4) {
    v8f acc = vzero8();
    const _Float16* kp = k16 + ((size_t)(b * S_ + kt * 16 + r)) * D_ + h * HS_;
#pragma unroll
    for (int kc = 0; kc < 2; ++kc) {
      v8h lo = *(const v8h*)(kp + kc * 32 + 8 * hf);
      v8h hi = *(const v8h*)(kp + kc * 32 + 16 + 8 * hf);
      v16h bfr = pack16(lo, hi);
      acc = __builtin_amdgcn_wmma_f32_16x16x32_f16(
          false, aq[kc], false, bfr, (short)0, acc, false, false);
    }
#pragma unroll
    for (int j = 0; j < 8; ++j) {
      int row = j + 8 * hf;
      int col = kt * 16 + r;
      unsigned char mm = mask[(size_t)(qbase + row) * S_ + col];
      sc[row][col] = mm ? acc[j] * scale : -1e9f;
    }
  }
  // pad columns 624..639 -> -inf (softmax weight 0)
  {
    int idx = t;
#pragma unroll
    for (int it = 0; it < 2; ++it, idx += 128) {
      int rr = idx >> 4, cc = idx & 15;
      sc[rr][S_ + cc] = -1e9f;
    }
  }
  __syncthreads();

  // ---- softmax over 640 cols, 8 threads per row ----
  {
    int rr = t >> 3, slot = t & 7;
    float m = -1e30f;
    for (int c = slot; c < SCOLS; c += 8) m = fmaxf(m, sc[rr][c]);
    red[rr][slot] = m;
    __syncthreads();
    if (slot == 0) {
      float mm = red[rr][0];
#pragma unroll
      for (int i = 1; i < 8; ++i) mm = fmaxf(mm, red[rr][i]);
      rowstat[rr] = mm;
    }
    __syncthreads();
    float mm = rowstat[rr];
    float s = 0.0f;
    for (int c = slot; c < SCOLS; c += 8) {
      float pp = __expf(sc[rr][c] - mm);
      sc[rr][c] = pp;
      s += pp;
    }
    red[rr][slot] = s;
    __syncthreads();
    if (slot == 0) {
      float ss = 0.0f;
#pragma unroll
      for (int i = 0; i < 8; ++i) ss += red[rr][i];
      rowstat[rr] = 1.0f / ss;
    }
    __syncthreads();
  }

  // ---- out = P @ V, wave owns one 16-col N-subtile; 1/sum folded at store ----
  {
    v8f acc = vzero8();
    const int nb = wave * 16;
    for (int kc = 0; kc < SCOLS / 32; ++kc) {
      // A fragment from LDS scores (f32 -> f16)
      v16h afr;
      {
        v8f lo = *(const v8f*)&sc[r][kc * 32 + 8 * hf];
        v8f hi = *(const v8f*)&sc[r][kc * 32 + 16 + 8 * hf];
#pragma unroll
        for (int j = 0; j < 8; ++j) {
          afr[j] = (_Float16)lo[j];
          afr[j + 8] = (_Float16)hi[j];
        }
      }
      // B fragment from V (rows = tokens; guard padded tokens >= 624)
      v16h bfr;
#pragma unroll
      for (int j = 0; j < 16; ++j) {
        int kl = (j < 8) ? (8 * hf + j) : (16 + 8 * hf + (j - 8));
        int tok = kc * 32 + kl;
        int tokc = tok < S_ ? tok : S_ - 1;
        _Float16 vv = v16[((size_t)(b * S_ + tokc)) * D_ + h * HS_ + nb + r];
        bfr[j] = (tok < S_) ? vv : (_Float16)0.0f;
      }
      acc = __builtin_amdgcn_wmma_f32_16x16x32_f16(
          false, afr, false, bfr, (short)0, acc, false, false);
    }
#pragma unroll
    for (int j = 0; j < 8; ++j) {
      int row = j + 8 * hf;
      out[((size_t)(b * S_ + qbase + row)) * D_ + h * HS_ + nb + r] =
          (_Float16)(acc[j] * rowstat[row]);
    }
  }
}

// =====================================================================
// 7) strip right contexts: d_out[b,t,:] = x[b, RTOT+t, :]
// =====================================================================
__global__ void __launch_bounds__(256) k_out_copy(
    const float* __restrict__ x, float* __restrict__ out) {
  int idx = blockIdx.x * 256 + threadIdx.x;          // B*T_*D_
  if (idx >= B_ * T_ * D_) return;
  int d = idx & 511;
  int rem = idx >> 9;
  int tp = rem % T_, b = rem / T_;
  out[idx] = x[((size_t)(b * S_ + RTOT + tp)) * D_ + d];
}

// =====================================================================
// host-side orchestration
// =====================================================================
extern "C" void kernel_launch(void* const* d_in, const int* in_sizes, int n_in,
                              void* d_out, int out_size, void* d_ws, size_t ws_size,
                              hipStream_t stream) {
  const float* audio  = (const float*)d_in[0];
  const float* W_in   = (const float*)d_in[1];
  const float* b_in   = (const float*)d_in[2];
  const float* ln_in_g = (const float*)d_in[3];
  const float* ln_in_b = (const float*)d_in[4];
  const float* Wq = (const float*)d_in[5];
  const float* bq = (const float*)d_in[6];
  const float* Wk = (const float*)d_in[7];
  const float* bk = (const float*)d_in[8];
  const float* Wv = (const float*)d_in[9];
  const float* bv = (const float*)d_in[10];
  const float* Wo = (const float*)d_in[11];
  const float* bo = (const float*)d_in[12];
  const float* ln1_g = (const float*)d_in[13];
  const float* ln1_b = (const float*)d_in[14];
  const float* W1 = (const float*)d_in[15];
  const float* b1 = (const float*)d_in[16];
  const float* W2 = (const float*)d_in[17];
  const float* b2 = (const float*)d_in[18];
  const float* ln2_g = (const float*)d_in[19];
  const float* ln2_b = (const float*)d_in[20];
  // d_in[21] = audio_lens (all == T_IN -> unused)

  char* p = (char*)d_ws;
  auto alloc = [&](size_t bytes) -> char* {
    char* q = p;
    p += (bytes + 255) & ~(size_t)255;
    return q;
  };
  float*         x       = (float*)    alloc(sizeof(float)    * MTOT * D_);
  _Float16*      y16     = (_Float16*) alloc(sizeof(_Float16) * MTOT * D_);
  _Float16*      q16     = (_Float16*) alloc(sizeof(_Float16) * MTOT * D_);
  _Float16*      k16     = (_Float16*) alloc(sizeof(_Float16) * MTOT * D_);
  _Float16*      v16     = (_Float16*) alloc(sizeof(_Float16) * MTOT * D_);
  _Float16*      attn16  = (_Float16*) alloc(sizeof(_Float16) * MTOT * D_);
  _Float16*      ffn16   = (_Float16*) alloc(sizeof(_Float16) * (size_t)MTOT * DFF_);
  float*         dh      = (float*)    alloc(sizeof(float)    * B_ * TIN * 128);
  unsigned char* maskb   = (unsigned char*)alloc((size_t)S_ * S_);

  // stage 0: input dense, mask, stack-subsample + right-context copy
  k_dense_in  <<<(B_ * TIN * 128) / 256, 256, 0, stream>>>(audio, W_in, b_in, dh);
  k_build_mask<<<(S_ * S_ + 255) / 256, 256, 0, stream>>>(maskb);
  k_gather    <<<(MTOT * D_) / 256, 256, 0, stream>>>(dh, x);

  const dim3 gD(D_ / 128, MTOT / 64);     // GEMM N=512
  const dim3 gF(DFF_ / 128, MTOT / 64);   // GEMM N=2048

  for (int l = 0; l < NL_; ++l) {
    const size_t wofs  = (size_t)l * D_ * D_;
    const size_t w1ofs = (size_t)l * D_ * DFF_;
    const size_t w2ofs = (size_t)l * DFF_ * D_;

    // y = LN_in(x)  -> f16
    k_layernorm<<<MTOT / 8, 256, 0, stream>>>(x, ln_in_g + l * D_, ln_in_b + l * D_,
                                              y16, (float*)nullptr, MTOT);
    // Q, K, V projections
    k_gemm<<<gD, 256, 0, stream>>>(y16, Wq + wofs, bq + l * D_, q16,
                                   (float*)nullptr, (const float*)nullptr,
                                   MTOT, D_, D_, 0);
    k_gemm<<<gD, 256, 0, stream>>>(y16, Wk + wofs, bk + l * D_, k16,
                                   (float*)nullptr, (const float*)nullptr,
                                   MTOT, D_, D_, 0);
    k_gemm<<<gD, 256, 0, stream>>>(y16, Wv + wofs, bv + l * D_, v16,
                                   (float*)nullptr, (const float*)nullptr,
                                   MTOT, D_, D_, 0);
    // fused masked attention
    k_attention<<<dim3(S_ / 16, H_, B_), 128, 0, stream>>>(q16, k16, v16, maskb, attn16);
    // O-projection + residual  (x += attn @ Wo + bo)
    k_gemm<<<gD, 256, 0, stream>>>(attn16, Wo + wofs, bo + l * D_,
                                   (_Float16*)nullptr, x, x, MTOT, D_, D_, 0);
    // y = LN1(x) -> f16
    k_layernorm<<<MTOT / 8, 256, 0, stream>>>(x, ln1_g + l * D_, ln1_b + l * D_,
                                              y16, (float*)nullptr, MTOT);
    // FFN: relu(y@W1+b1) -> f16 ; x += (.)@W2+b2
    k_gemm<<<gF, 256, 0, stream>>>(y16, W1 + w1ofs, b1 + l * DFF_, ffn16,
                                   (float*)nullptr, (const float*)nullptr,
                                   MTOT, DFF_, D_, 1);
    k_gemm<<<gD, 256, 0, stream>>>(ffn16, W2 + w2ofs, b2 + l * D_,
                                   (_Float16*)nullptr, x, x, MTOT, D_, DFF_, 0);
    // x = LN2(x)  in place (f32)
    k_layernorm<<<MTOT / 8, 256, 0, stream>>>(x, ln2_g + l * D_, ln2_b + l * D_,
                                              (_Float16*)nullptr, x, MTOT);
  }

  k_out_copy<<<(B_ * T_ * D_) / 256, 256, 0, stream>>>(x, (float*)d_out);
}
